// Conv4D_68281390072023
// MI455X (gfx1250) — matrix operand order repaired and checked by reading, not compiled, not run on hardware
//
#include <hip/hip_runtime.h>

// ---- problem constants (match reference) ----
#define CIN   64
#define COUTC 128
#define DSP   24
#define SPAT  (DSP * DSP * DSP)      // 13824
#define BATCH 4
#define NPOS  (BATCH * SPAT)         // 55296
#define KDIM  (CIN * 27)             // 1728

typedef __attribute__((ext_vector_type(16))) __bf16        v16bf;
typedef __attribute__((ext_vector_type(8)))  __bf16        v8bf;
typedef __attribute__((ext_vector_type(8)))  float         v8f;
typedef __attribute__((ext_vector_type(4)))  unsigned int  v4u;
typedef int v4i __attribute__((vector_size(16)));   // matches builtin param type

// ---- CDNA5 async global->LDS copy (guarded; falls back to reg-mediated copy) ----
#if __has_builtin(__builtin_amdgcn_global_load_async_to_lds_b128)
#define HAVE_ASYNC 1
#else
#define HAVE_ASYNC 0
#endif

#if HAVE_ASYNC
#if __has_builtin(__builtin_amdgcn_s_wait_asynccnt)
#define WAIT_ASYNC() __builtin_amdgcn_s_wait_asynccnt(0)
#else
#define WAIT_ASYNC() asm volatile("s_wait_asynccnt 0x0" ::: "memory")
#endif
#else
#define WAIT_ASYNC() ((void)0)
#endif

static __device__ __forceinline__ void cp16(const unsigned short* g, unsigned short* l) {
#if HAVE_ASYNC
    __builtin_amdgcn_global_load_async_to_lds_b128(
        (__attribute__((address_space(1))) v4i*)g,
        (__attribute__((address_space(3))) v4i*)l, 0, 0);
#else
    *(v4u*)l = *(const v4u*)g;
#endif
}

// round-to-nearest fp32 -> bf16 pair pack (a -> low half)
static __device__ __forceinline__ unsigned int pack_bf16x2(float a, float b) {
    unsigned int ua = __builtin_bit_cast(unsigned int, a);
    unsigned int ub = __builtin_bit_cast(unsigned int, b);
    ua = ua + 0x7FFFu + ((ua >> 16) & 1u);
    ub = ub + 0x7FFFu + ((ub >> 16) & 1u);
    return (ua >> 16) | (ub & 0xFFFF0000u);
}

// ============================================================================
// Pre-pass 1: x (4,64,24^3) f32 -> xT (4,24^3,64) bf16  (cin innermost)
// ============================================================================
__global__ __launch_bounds__(256) void xpose_bf16(const float* __restrict__ X,
                                                  unsigned short* __restrict__ xT) {
    const int posIdx = blockIdx.x * 256 + threadIdx.x;   // 0..55295
    const int cip    = blockIdx.y;                       // 0..31 cin pairs
    const int pb     = posIdx / SPAT;
    const int s      = posIdx - pb * SPAT;
    const float* xp  = X + ((size_t)pb * CIN + 2 * cip) * SPAT + s;
    const unsigned int pk = pack_bf16x2(xp[0], xp[SPAT]);
    *(unsigned int*)&xT[(size_t)posIdx * CIN + 2 * cip] = pk;
}

// ============================================================================
// Pre-pass 2: w (128, cin*27) f32 [ci*27+t] -> w2 (128, 27, 64) bf16 [t*64+ci]
// ============================================================================
__global__ __launch_bounds__(256) void wconv_bf16(const float* __restrict__ W,
                                                  unsigned short* __restrict__ w2,
                                                  unsigned int* __restrict__ zblk) {
    const int gid  = blockIdx.x * 256 + threadIdx.x;     // 0..110591
    const int oid  = gid * 2;
    const int cout = oid / KDIM;
    const int r    = oid - cout * KDIM;
    const int t    = r >> 6;
    const int ci   = r & 63;
    const float f0 = W[(size_t)cout * KDIM + ci * 27 + t];
    const float f1 = W[(size_t)cout * KDIM + (ci + 1) * 27 + t];
    *(unsigned int*)&w2[oid] = pack_bf16x2(f0, f1);
    if (blockIdx.x == 0 && threadIdx.x < 32) zblk[threadIdx.x] = 0u;
}

// ============================================================================
// Main implicit-GEMM kernel, all-bf16 data path, async double-buffered LDS.
// Block: 64 cout x 128 pos, 8 waves (waveM 0..3 x waveN 0..1), wave = 16x64.
// Tap loop: 27 steps of K=64 (8 WMMA/wave/step). LDS rows padded to 144B.
// ============================================================================
#define LDA_ROW 72   // ushorts per A row (64 data + 8 pad)
#define LDB_ROW 72   // ushorts per B row

__global__ __launch_bounds__(256) void conv3d_wmma_async(
    const unsigned short* __restrict__ xT,   // (4*13824, 64) bf16
    const unsigned short* __restrict__ w2,   // (128, 27, 64) bf16
    const unsigned short* __restrict__ zb,   // 128B zeros
    float* __restrict__ O)                   // (4,128,24,24,24) f32
{
    __shared__ __align__(16) unsigned short ldsA[2][64 * LDA_ROW];   // 2 x 9 KB
    __shared__ __align__(16) unsigned short ldsB[2][128 * LDB_ROW];  // 2 x 18 KB

    const int tid    = threadIdx.x;
    const int bx     = blockIdx.x;
    const int m0     = (bx & 1) * 64;
    const int n0     = (bx >> 1) * 128;

    // staging roles: A = 32B/thread (2x b128), B = 64B/thread (4x b128)
    const int am_  = tid >> 2;        // 0..63   A row
    const int ach  = tid & 3;         // 0..3    32B chunk within A row
    const int col  = tid & 127;       // B column (position)
    const int half = tid >> 7;        // 0/1     64B half of B row

    // decompose this thread's position once
    const int p    = n0 + col;
    const int pb   = p / SPAT;
    const int prem = p - pb * SPAT;
    const int pz   = prem / 576;
    const int py   = (prem - pz * 576) / 24;
    const int px   = prem - pz * 576 - py * 24;

    const unsigned short* aSrc = w2 + (size_t)(m0 + am_) * KDIM + ach * 16;
    const unsigned short* bPos = xT + (size_t)pb * SPAT * CIN + half * 32;

    // wave roles
    const int lane  = tid & 31;
    const int grp   = lane >> 4;      // K-half select in fragments
    const int mrow  = lane & 15;      // M row (A) / N col (B, C)
    const int waveM = (tid >> 5) & 3; // cout subtile (16)
    const int waveN = tid >> 7;       // pos subtile (64)

    v8f acc[4] = {{}, {}, {}, {}};

    // issue async copies for tap t into buffer t&1
    auto issue = [&](int t) {
        const int kd = t / 9;
        const int kh = (t - kd * 9) / 3;
        const int kw = t - kd * 9 - kh * 3;
        const int zi = pz + kd - 1, yi = py + kh - 1, xi = px + kw - 1;
        const bool valid = ((unsigned)zi < (unsigned)DSP) &
                           ((unsigned)yi < (unsigned)DSP) &
                           ((unsigned)xi < (unsigned)DSP);
        unsigned short* A = &ldsA[t & 1][0];
        unsigned short* B = &ldsB[t & 1][0];
        // A: 64 rows x 128B, fully contiguous per row in w2
        const unsigned short* as = aSrc + t * 64;
        unsigned short* ad = &A[am_ * LDA_ROW + ach * 16];
        cp16(as,     ad);
        cp16(as + 8, ad + 8);
        // B: one 64B half-row per thread; OOB columns copy from the zero block
        const int spat = zi * 576 + yi * 24 + xi;
        const unsigned short* bs = valid ? (bPos + (size_t)spat * CIN) : zb;
        unsigned short* bd = &B[col * LDB_ROW + half * 32];
        cp16(bs,      bd);
        cp16(bs + 8,  bd + 8);
        cp16(bs + 16, bd + 16);
        cp16(bs + 24, bd + 24);
    };

    issue(0);
    WAIT_ASYNC();
    __syncthreads();

    for (int t = 0; t < 27; ++t) {
        if (t < 26) issue(t + 1);   // overlaps with compute below

        const unsigned short* A = &ldsA[t & 1][0];
        const unsigned short* B = &ldsB[t & 1][0];
#pragma unroll
        for (int c2 = 0; c2 < 2; ++c2) {
            // A fragment: K = grp*16 + h, contiguous 32B (two 16B reads)
            union { v16bf v; v8bf h[2]; } au;
            const int ab = (waveM * 16 + mrow) * LDA_ROW + c2 * 32;
            au.h[0] = *(const v8bf*)&A[ab +     grp * 16];
            au.h[1] = *(const v8bf*)&A[ab + 8 + grp * 16];
#pragma unroll
            for (int sN = 0; sN < 4; ++sN) {
                // B fragment: K = grp*16 + h, contiguous 32B (two 16B reads)
                union { v16bf v; v8bf h[2]; } bu;
                const int bb = (waveN * 64 + sN * 16 + mrow) * LDB_ROW
                             + c2 * 32 + grp * 16;
                bu.h[0] = *(const v8bf*)&B[bb];
                bu.h[1] = *(const v8bf*)&B[bb + 8];
                acc[sN] = __builtin_amdgcn_wmma_f32_16x16x32_bf16(
                              false, au.v, false, bu.v, (short)0, acc[sN],
                              false, false);
            }
        }
        WAIT_ASYNC();       // copies for t+1 done (overlapped with the 8 WMMAs)
        __syncthreads();    // everyone finished reading buffer t
    }

    // write out: acc VGPR r holds M = grp*8 + r, N = mrow
#pragma unroll
    for (int s = 0; s < 4; ++s) {
        const int pp   = n0 + waveN * 64 + s * 16 + mrow;
        const int ob   = pp / SPAT;
        const int orem = pp - ob * SPAT;
        float* op = O + (size_t)ob * (COUTC * SPAT)
                      + (size_t)(m0 + waveM * 16 + grp * 8) * SPAT + orem;
#pragma unroll
        for (int r = 0; r < 8; ++r)
            op[(size_t)r * SPAT] = acc[s][r];        // coalesced across lanes
    }
}

// ============================================================================
// Fallback fused kernel (used only if ws_size is too small): converts in-loop.
// ============================================================================
__global__ __launch_bounds__(256) void conv3d_fused_fallback(
    const float* __restrict__ X, const float* __restrict__ W,
    float* __restrict__ O)
{
    __shared__ __align__(32) unsigned short ldsA[64 * 32];
    __shared__ __align__(32) unsigned short ldsB[128 * 32];

    const int tid = threadIdx.x, bx = blockIdx.x;
    const int m0 = (bx & 1) * 64, n0 = (bx >> 1) * 128;
    const int colLoc = tid & 127, rowHalf = tid >> 7;
    const int am = tid & 63, akb = (tid >> 6) << 3;
    const int p = n0 + colLoc, pb = p / SPAT, prem = p - pb * SPAT;
    const int pz = prem / 576, py = (prem - pz * 576) / 24, px = prem - pz * 576 - py * 24;
    const int lane = tid & 31, grp = lane >> 4, mrow = lane & 15;
    const int waveM = (tid >> 5) & 3, waveN = tid >> 7;
    v8f acc[4] = {{}, {}, {}, {}};

    for (int t = 0; t < 27; ++t) {
        const int kd = t / 9, kh = (t - kd * 9) / 3, kw = t - kd * 9 - kh * 3;
        const int zi = pz + kd - 1, yi = py + kh - 1, xi = px + kw - 1;
        const bool valid = ((unsigned)zi < (unsigned)DSP) & ((unsigned)yi < (unsigned)DSP) &
                           ((unsigned)xi < (unsigned)DSP);
        const int spat = valid ? (zi * 576 + yi * 24 + xi) : 0;
        for (int c2 = 0; c2 < 2; ++c2) {
            __syncthreads();
            {
                const float* wp = W + (size_t)(m0 + am) * KDIM + (size_t)(c2 * 32 + akb) * 27 + t;
                v4u pk;
#pragma unroll
                for (int j = 0; j < 4; ++j)
                    pk[j] = pack_bf16x2(wp[(2 * j) * 27], wp[(2 * j + 1) * 27]);
                *(v4u*)&ldsA[am * 32 + akb] = pk;
            }
            {
                const float* xp = X + ((size_t)pb * CIN + c2 * 32 + rowHalf * 16) * SPAT + spat;
                v4u pk0, pk1;
#pragma unroll
                for (int j = 0; j < 4; ++j) {
                    float f0 = xp[(2 * j) * SPAT], f1 = xp[(2 * j + 1) * SPAT];
                    if (!valid) { f0 = 0.f; f1 = 0.f; }
                    pk0[j] = pack_bf16x2(f0, f1);
                }
#pragma unroll
                for (int j = 0; j < 4; ++j) {
                    float f0 = xp[(8 + 2 * j) * SPAT], f1 = xp[(9 + 2 * j) * SPAT];
                    if (!valid) { f0 = 0.f; f1 = 0.f; }
                    pk1[j] = pack_bf16x2(f0, f1);
                }
                unsigned short* bp = &ldsB[colLoc * 32 + rowHalf * 16];
                *(v4u*)bp = pk0;  *(v4u*)(bp + 8) = pk1;
            }
            __syncthreads();
            union { v16bf v; v8bf h[2]; } au;
            const int arow = (waveM * 16 + mrow) * 32;
            au.h[0] = *(const v8bf*)&ldsA[arow + grp * 16];
            au.h[1] = *(const v8bf*)&ldsA[arow + 8 + grp * 16];
#pragma unroll
            for (int sN = 0; sN < 4; ++sN) {
                union { v16bf v; v8bf h[2]; } bu;
                const int bb = (waveN * 64 + sN * 16 + mrow) * 32 + grp * 16;
                bu.h[0] = *(const v8bf*)&ldsB[bb];
                bu.h[1] = *(const v8bf*)&ldsB[bb + 8];
                acc[sN] = __builtin_amdgcn_wmma_f32_16x16x32_bf16(
                              false, au.v, false, bu.v, (short)0, acc[sN], false, false);
            }
        }
    }
#pragma unroll
    for (int s = 0; s < 4; ++s) {
        const int pp = n0 + waveN * 64 + s * 16 + mrow;
        const int ob = pp / SPAT, orem = pp - ob * SPAT;
        float* op = O + (size_t)ob * (COUTC * SPAT)
                      + (size_t)(m0 + waveM * 16 + grp * 8) * SPAT + orem;
#pragma unroll
        for (int r = 0; r < 8; ++r) op[(size_t)r * SPAT] = acc[s][r];
    }
}

extern "C" void kernel_launch(void* const* d_in, const int* in_sizes, int n_in,
                              void* d_out, int out_size, void* d_ws, size_t ws_size,
                              hipStream_t stream) {
    (void)in_sizes; (void)n_in; (void)out_size;
    const float* x = (const float*)d_in[0];   // (4,64,24,24,24) f32
    const float* w = (const float*)d_in[1];   // (1,128,1728)    f32
    float* out = (float*)d_out;

    const size_t xT_bytes = (size_t)NPOS * CIN * 2;            // 7,077,888
    const size_t w2_bytes = (size_t)COUTC * KDIM * 2;          //   442,368
    const size_t need     = xT_bytes + w2_bytes + 128;
    const int    blocks   = (COUTC / 64) * (NPOS / 128);       // 864

    if (ws_size >= need) {
        unsigned char* ws = (unsigned char*)d_ws;
        unsigned short* xT = (unsigned short*)ws;
        unsigned short* w2 = (unsigned short*)(ws + xT_bytes);
        unsigned short* zb = (unsigned short*)(ws + xT_bytes + w2_bytes);

        xpose_bf16<<<dim3(NPOS / 256, CIN / 2), 256, 0, stream>>>(x, xT);
        wconv_bf16<<<(COUTC * KDIM / 2 + 255) / 256, 256, 0, stream>>>(
            w, w2, (unsigned int*)zb);
        conv3d_wmma_async<<<blocks, 256, 0, stream>>>(xT, w2, zb, out);
    } else {
        conv3d_fused_fallback<<<blocks, 256, 0, stream>>>(x, w, out);
    }
}
